// CoordinateLinear_56495999811946
// MI455X (gfx1250) — matrix operand order repaired and checked by reading, not compiled, not run on hardware
//
#include <hip/hip_runtime.h>

typedef __attribute__((ext_vector_type(4)))  __bf16 v4bf;
typedef __attribute__((ext_vector_type(8)))  __bf16 v8bf;
typedef __attribute__((ext_vector_type(16))) __bf16 v16bf;
typedef __attribute__((ext_vector_type(8)))  float  v8f;
typedef __attribute__((ext_vector_type(4)))  float  f32x4;

#define BATCH 16384
#define IN_F  1024
#define OUT_F 1024

#define BM 128
#define BN 128
#define BK 32
#define KPAD 40            // BK + 8 bf16 pad -> 80B row stride, 16B-aligned fragments
#define NSTEP (IN_F / BK)  // 32 K-steps

__device__ __forceinline__ void split_bf16(float x, __bf16& hi, __bf16& lo) {
    hi = (__bf16)x;
    lo = (__bf16)(x - (float)hi);
}

__device__ __forceinline__ v16bf cat8(v8bf a, v8bf b) {
    return __builtin_shufflevector(a, b, 0,1,2,3,4,5,6,7,8,9,10,11,12,13,14,15);
}

__global__ __launch_bounds__(256) void coord_linear_wmma(
    const float* __restrict__ X,     // [IN_F][BATCH]
    const float* __restrict__ W,     // [OUT_F][IN_F]
    const float* __restrict__ Bias,  // [OUT_F]
    float* __restrict__ Out)         // [OUT_F][BATCH]
{
    // Double-buffered split-bf16 tiles: 4 arrays x 2 bufs x 128x40 x 2B = 80 KB.
    __shared__ __align__(16) __bf16 sWh[2][BM * KPAD];
    __shared__ __align__(16) __bf16 sWl[2][BM * KPAD];
    __shared__ __align__(16) __bf16 sXh[2][BN * KPAD];   // transposed: [n][k]
    __shared__ __align__(16) __bf16 sXl[2][BN * KPAD];

    const int tid  = threadIdx.x;
    const int lane = tid & 31;
    const int wave = tid >> 5;
    const int wm   = wave & 1;    // 0..1 : 64 M-rows each
    const int wn   = wave >> 1;   // 0..3 : 32 N-cols each
    const int g    = lane >> 4;   // lane group
    const int l15  = lane & 15;

    const int nblk = blockIdx.x * BN;   // batch offset
    const int mblk = blockIdx.y * BM;   // out-feature offset

    // Per-thread staging coordinates (4 float4 chunks for each of W and X).
    // W tile: 128 rows x 32 k -> 8 float4/row.  X tile: 32 k x 128 n -> 32 float4/row.
    int wrow[4], wcol4[4], xrow[4], xcol4[4];
#pragma unroll
    for (int i = 0; i < 4; ++i) {
        const int c = tid + i * 256;
        wrow[i]  = c >> 3;
        wcol4[i] = (c & 7) * 4;
        xrow[i]  = c >> 5;
        xcol4[i] = (c & 31) * 4;
    }

    v8f acc[4][2];
#pragma unroll
    for (int i = 0; i < 4; ++i)
#pragma unroll
        for (int j = 0; j < 2; ++j)
            acc[i][j] = (v8f){0.f,0.f,0.f,0.f,0.f,0.f,0.f,0.f};

    f32x4 wreg[4], xreg[4];

    // ---- prologue: fetch K-tile 0 and stage it into buffer 0 ----
#pragma unroll
    for (int i = 0; i < 4; ++i) {
        wreg[i] = *(const f32x4*)(W + (size_t)(mblk + wrow[i]) * IN_F + wcol4[i]);
        xreg[i] = *(const f32x4*)(X + (size_t)xrow[i] * BATCH + nblk + xcol4[i]);
    }
#pragma unroll
    for (int i = 0; i < 4; ++i) {
        v4bf h4, l4;
#pragma unroll
        for (int j = 0; j < 4; ++j) {
            __bf16 h, l;
            split_bf16(wreg[i][j], h, l);
            h4[j] = h; l4[j] = l;
        }
        *(v4bf*)&sWh[0][wrow[i] * KPAD + wcol4[i]] = h4;   // packed 8B LDS store
        *(v4bf*)&sWl[0][wrow[i] * KPAD + wcol4[i]] = l4;
#pragma unroll
        for (int j = 0; j < 4; ++j) {                      // transpose scatter
            __bf16 h, l;
            split_bf16(xreg[i][j], h, l);
            sXh[0][(xcol4[i] + j) * KPAD + xrow[i]] = h;
            sXl[0][(xcol4[i] + j) * KPAD + xrow[i]] = l;
        }
    }
    __syncthreads();

    for (int step = 0; step < NSTEP; ++step) {
        const int cur  = step & 1;
        const int nxt  = cur ^ 1;
        const bool more = (step + 1) < NSTEP;

        // ---- prefetch next K-tile into registers (overlaps with WMMAs) ----
        if (more) {
            const int k0n = (step + 1) * BK;
#pragma unroll
            for (int i = 0; i < 4; ++i) {
                wreg[i] = *(const f32x4*)(W + (size_t)(mblk + wrow[i]) * IN_F + k0n + wcol4[i]);
                xreg[i] = *(const f32x4*)(X + (size_t)(k0n + xrow[i]) * BATCH + nblk + xcol4[i]);
            }
        }

        // ---- compute from buf[cur] ----
        // B 32x16 layout: lane = column n; lanes 0-15: K=0..15, lanes 16-31: K=16..31.
        v16bf bh[2], bl[2];
#pragma unroll
        for (int tn = 0; tn < 2; ++tn) {
            const int n    = wn * 32 + tn * 16 + l15;
            const int base = n * KPAD + g * 16;
            bh[tn] = cat8(*(const v8bf*)(&sXh[cur][base]), *(const v8bf*)(&sXh[cur][base + 8]));
            bl[tn] = cat8(*(const v8bf*)(&sXl[cur][base]), *(const v8bf*)(&sXl[cur][base + 8]));
        }
        // A 16x32 layout: lane = row m; lanes 0-15: K=0..15, lanes 16-31: K=16..31 (matches B).
#pragma unroll
        for (int tm = 0; tm < 4; ++tm) {
            const int m    = wm * 64 + tm * 16 + l15;
            const int base = m * KPAD + g * 16;
            const v16bf ah = cat8(*(const v8bf*)(&sWh[cur][base]), *(const v8bf*)(&sWh[cur][base + 8]));
            const v16bf al = cat8(*(const v8bf*)(&sWl[cur][base]), *(const v8bf*)(&sWl[cur][base + 8]));
#pragma unroll
            for (int tn = 0; tn < 2; ++tn) {
                // bf16x3 split: hi*hi + hi*lo + lo*hi, f32 accumulate
                acc[tm][tn] = __builtin_amdgcn_wmma_f32_16x16x32_bf16(
                    false, ah, false, bh[tn], (short)0, acc[tm][tn], false, false);
                acc[tm][tn] = __builtin_amdgcn_wmma_f32_16x16x32_bf16(
                    false, ah, false, bl[tn], (short)0, acc[tm][tn], false, false);
                acc[tm][tn] = __builtin_amdgcn_wmma_f32_16x16x32_bf16(
                    false, al, false, bh[tn], (short)0, acc[tm][tn], false, false);
            }
        }

        // ---- stage prefetched tile into buf[nxt] (other buffer: no WAR on buf[cur]) ----
        if (more) {
#pragma unroll
            for (int i = 0; i < 4; ++i) {
                v4bf h4, l4;
#pragma unroll
                for (int j = 0; j < 4; ++j) {
                    __bf16 h, l;
                    split_bf16(wreg[i][j], h, l);
                    h4[j] = h; l4[j] = l;
                }
                *(v4bf*)&sWh[nxt][wrow[i] * KPAD + wcol4[i]] = h4;
                *(v4bf*)&sWl[nxt][wrow[i] * KPAD + wcol4[i]] = l4;
#pragma unroll
                for (int j = 0; j < 4; ++j) {
                    __bf16 h, l;
                    split_bf16(xreg[i][j], h, l);
                    sXh[nxt][(xcol4[i] + j) * KPAD + xrow[i]] = h;
                    sXl[nxt][(xcol4[i] + j) * KPAD + xrow[i]] = l;
                }
            }
        }
        __syncthreads();   // one barrier per K-step
    }

    // ---- epilogue: bias add + streaming stores ----
    // C/D layout: VGPR v -> M = v (lanes 0-15) or v+8 (lanes 16-31); N = lane&15.
#pragma unroll
    for (int tm = 0; tm < 4; ++tm) {
#pragma unroll
        for (int v = 0; v < 8; ++v) {
            const int o  = mblk + wm * 64 + tm * 16 + g * 8 + v;
            const float bv = Bias[o];
#pragma unroll
            for (int tn = 0; tn < 2; ++tn) {
                const int b = nblk + wn * 32 + tn * 16 + l15;
                __builtin_nontemporal_store(acc[tm][tn][v] + bv,
                                            Out + (size_t)o * BATCH + b);
            }
        }
    }
}

extern "C" void kernel_launch(void* const* d_in, const int* in_sizes, int n_in,
                              void* d_out, int out_size, void* d_ws, size_t ws_size,
                              hipStream_t stream) {
    const float* x    = (const float*)d_in[0];  // [IN_F, BATCH]
    const float* w    = (const float*)d_in[1];  // [OUT_F, IN_F]
    const float* bias = (const float*)d_in[2];  // [OUT_F]
    float* out        = (float*)d_out;          // [OUT_F, BATCH]

    dim3 grid(BATCH / BN, OUT_F / BM);          // 128 x 8
    coord_linear_wmma<<<grid, dim3(256), 0, stream>>>(x, w, bias, out);
}
